// SpatialAttention_85856396247108
// MI455X (gfx1250) — compile-verified, matrix operand order checked by reading, not run on hardware
//
#include <hip/hip_runtime.h>
#include <hip/hip_bf16.h>

// B=16, N=4096, C=512, S=8, H=8, hd=64
#define Bb 16
#define Nn 4096
#define Cc 512
#define Ss 8
#define Hh 8
#define HD 64

typedef __attribute__((ext_vector_type(16))) _Float16 v16h;
typedef __attribute__((ext_vector_type(8)))  float    v8f;

__device__ __forceinline__ void put4(v16h& v, int base, float4 f) {
  v[base + 0] = (_Float16)f.x;
  v[base + 1] = (_Float16)f.y;
  v[base + 2] = (_Float16)f.z;
  v[base + 3] = (_Float16)f.w;
}

// ---------------------------------------------------------------------------
// K1: sa_lin = x @ w_sum^T + b_sum, stored flat as A[b][n*8+s].
// WMMA f16: wave computes 16(row) x 16(col, S padded to 16) f32 tile, K=512.
// Padded B columns (8..15) alias real rows: finite garbage that only lands in
// D columns we never store — no masking needed at all.
// ---------------------------------------------------------------------------
__global__ __launch_bounds__(256) void k1_spaceattn(
    const float* __restrict__ x, const float* __restrict__ w_sum,
    const float* __restrict__ b_sum, float* __restrict__ A)
{
  const int wave = threadIdx.x >> 5;
  const int lane = threadIdx.x & 31;
  const int half = lane >> 4;
  const int l16  = lane & 15;

  const int rowTile = blockIdx.x * 8 + wave;   // 4096 tiles of 16 rows
  const int row0 = rowTile * 16;

  const float* xrow = x + (size_t)(row0 + l16) * Cc;   // A row M = l%16
  const bool nvalid = l16 < Ss;                        // B col (S padded)
  const float* wrow = w_sum + (size_t)(l16 & 7) * Cc;  // always in-bounds

  v8f acc = {};
  for (int kk = 0; kk < Cc / 32; ++kk) {
    const int kbase = kk * 32 + half * 8;
    __builtin_prefetch(xrow + kbase + 128, 0, 3);

    float4 a0 = *(const float4*)(xrow + kbase);
    float4 a1 = *(const float4*)(xrow + kbase + 4);
    float4 a2 = *(const float4*)(xrow + kbase + 16);
    float4 a3 = *(const float4*)(xrow + kbase + 20);
    float4 b0 = *(const float4*)(wrow + kbase);
    float4 b1 = *(const float4*)(wrow + kbase + 4);
    float4 b2 = *(const float4*)(wrow + kbase + 16);
    float4 b3 = *(const float4*)(wrow + kbase + 20);

    v16h av, bv;
    put4(av, 0, a0); put4(av, 4, a1); put4(av, 8, a2); put4(av, 12, a3);
    put4(bv, 0, b0); put4(bv, 4, b1); put4(bv, 8, b2); put4(bv, 12, b3);

    acc = __builtin_amdgcn_wmma_f32_16x16x32_f16(
        false, av, false, bv, (short)0, acc, false, false);
  }

  // D element r -> (M = r + 8*half, N = l16). Keep N < 8 (valid S cols).
  if (nvalid) {
    const float bias = b_sum[l16];
#pragma unroll
    for (int r = 0; r < 8; ++r) {
      const int grow = row0 + r + 8 * half;   // global row in [0, B*N)
      const int b  = grow >> 12;
      const int nl = grow & (Nn - 1);
      A[(size_t)b * (Nn * Ss) + (size_t)nl * Ss + l16] = acc[r] + bias;
    }
  }
}

// ---------------------------------------------------------------------------
// K2: softmax over each contiguous 4096-segment of A -> P. 128 segments.
// b128 loads/stores throughout.
// ---------------------------------------------------------------------------
__global__ __launch_bounds__(256) void k2_softmax(
    const float* __restrict__ A, float* __restrict__ P)
{
  const int seg = blockIdx.x;
  const float4* a4 = (const float4*)(A + (size_t)seg * 4096);
  float4* p4 = (float4*)(P + (size_t)seg * 4096);
  const int t = threadIdx.x, lane = t & 31, wave = t >> 5;

  float4 v[4];
  float m = -3.4e38f;
#pragma unroll
  for (int i = 0; i < 4; ++i) {
    v[i] = a4[t + 256 * i];
    m = fmaxf(m, fmaxf(fmaxf(v[i].x, v[i].y), fmaxf(v[i].z, v[i].w)));
  }
  for (int off = 16; off > 0; off >>= 1) m = fmaxf(m, __shfl_xor(m, off, 32));

  __shared__ float redm[8], redz[8];
  if (lane == 0) redm[wave] = m;
  __syncthreads();
  if (t == 0) {
    float mm = redm[0];
    for (int i = 1; i < 8; ++i) mm = fmaxf(mm, redm[i]);
    redm[0] = mm;
  }
  __syncthreads();
  m = redm[0];

  float z = 0.0f;
#pragma unroll
  for (int i = 0; i < 4; ++i) {
    v[i].x = __expf(v[i].x - m); v[i].y = __expf(v[i].y - m);
    v[i].z = __expf(v[i].z - m); v[i].w = __expf(v[i].w - m);
    z += v[i].x + v[i].y + v[i].z + v[i].w;
  }
  for (int off = 16; off > 0; off >>= 1) z += __shfl_xor(z, off, 32);
  if (lane == 0) redz[wave] = z;
  __syncthreads();
  if (t == 0) {
    float s = 0.0f;
    for (int i = 0; i < 8; ++i) s += redz[i];
    redz[0] = s;
  }
  __syncthreads();
  const float inv = 1.0f / redz[0];
#pragma unroll
  for (int i = 0; i < 4; ++i) {
    float4 o = v[i];
    o.x *= inv; o.y *= inv; o.z *= inv; o.w *= inv;
    p4[t + 256 * i] = o;
  }
}

// ---------------------------------------------------------------------------
// K3: tokens[b,s,c] = max_n P[b][s*4096+n] * Xf[b][c*4096+n].
// Block = (b, 32-c tile); thread owns one (s, c_local) accumulator.
// Global staging uses b128 loads; compute tile keeps 257-pad (conflict-free).
// ---------------------------------------------------------------------------
__global__ __launch_bounds__(256) void k3_tokens(
    const float* __restrict__ x, const float* __restrict__ P,
    float* __restrict__ tokens)
{
  const int b  = blockIdx.x >> 4;
  const int ct = blockIdx.x & 15;
  const int t = threadIdx.x, lane = t & 31, s = t >> 5;

  __shared__ float xs[32][257];   // padded: conflict-free in the hot loop
  __shared__ float pt[8][256];

  const float* Xf = x + (size_t)b * Nn * Cc;        // flat (C*N) view
  const float4* Pb4 = (const float4*)(P + (size_t)b * (Nn * Ss));
  const int c0 = ct * 32;
  const int cr = t >> 6;          // 0..3: c sub-row for staging
  const int q  = t & 63;          // float4 index along n

  float acc = -3.4e38f;
  for (int chunk = 0; chunk < 16; ++chunk) {
    const int n0 = chunk * 256;
#pragma unroll
    for (int c4 = 0; c4 < 8; ++c4) {
      const int c = c4 * 4 + cr;
      const float4* src =
          (const float4*)(Xf + (size_t)(c0 + c) * 4096 + n0) + q;
      __builtin_prefetch(src + 64, 0, 3);
      float4 val = *src;
      xs[c][q * 4 + 0] = val.x;
      xs[c][q * 4 + 1] = val.y;
      xs[c][q * 4 + 2] = val.z;
      xs[c][q * 4 + 3] = val.w;
    }
#pragma unroll
    for (int rr = 0; rr < 2; ++rr) {
      const int idx = rr * 256 + t;     // 512 float4s total
      const int r = idx >> 6, qq = idx & 63;
      *(float4*)&pt[r][qq * 4] = Pb4[r * 1024 + (n0 >> 2) + qq];
    }
    __syncthreads();
#pragma unroll 8
    for (int j = 0; j < 256; ++j)
      acc = fmaxf(acc, pt[s][j] * xs[lane][j]);
    __syncthreads();
  }
  tokens[((size_t)b * Ss + s) * Cc + c0 + lane] = acc;
}

// ---------------------------------------------------------------------------
// K4: per-batch. WMMA qkv = tokens(8->16 pad, 512) @ w_qkv^T (512x1536) -> LDS,
// then 8-token attention (H=8, hd=64), softmax, attn@v, cross-mix over S.
// Padded A rows (8..15) alias rows 0..7: garbage lands only in D rows we
// never store (half==1 lanes), so no masking is required.
// ---------------------------------------------------------------------------
__global__ __launch_bounds__(256) void k4_attn(
    const float* __restrict__ tokens, const float* __restrict__ w_qkv,
    const float* __restrict__ w_cross, float* __restrict__ asp2)
{
  const int b = blockIdx.x;
  __shared__ float qkv[8][1536];      // 48 KB: [s][q|k|v]
  __shared__ float scores[8][8][8];   // [h][i][j]
  const int t = threadIdx.x, lane = t & 31, wave = t >> 5;
  const int half = lane >> 4, l16 = lane & 15;

  const float* tok = tokens + (size_t)b * Ss * Cc;
  const float* arow = tok + (size_t)(l16 & 7) * Cc;   // always in-bounds

  for (int tt = 0; tt < 12; ++tt) {
    const int nt = wave * 12 + tt;                    // 96 N-tiles of 16
    const float* brow = w_qkv + (size_t)(nt * 16 + l16) * Cc;
    v8f acc = {};
    for (int kk = 0; kk < Cc / 32; ++kk) {
      const int kbase = kk * 32 + half * 8;
      float4 a0 = *(const float4*)(arow + kbase);
      float4 a1 = *(const float4*)(arow + kbase + 4);
      float4 a2 = *(const float4*)(arow + kbase + 16);
      float4 a3 = *(const float4*)(arow + kbase + 20);
      float4 b0 = *(const float4*)(brow + kbase);
      float4 b1 = *(const float4*)(brow + kbase + 4);
      float4 b2 = *(const float4*)(brow + kbase + 16);
      float4 b3 = *(const float4*)(brow + kbase + 20);

      v16h av, bv;
      put4(av, 0, a0); put4(av, 4, a1); put4(av, 8, a2); put4(av, 12, a3);
      put4(bv, 0, b0); put4(bv, 4, b1); put4(bv, 8, b2); put4(bv, 12, b3);

      acc = __builtin_amdgcn_wmma_f32_16x16x32_f16(
          false, av, false, bv, (short)0, acc, false, false);
    }
    if (half == 0) {                  // rows M = r (0..7) valid
#pragma unroll
      for (int r = 0; r < 8; ++r) qkv[r][nt * 16 + l16] = acc[r];
    }
  }
  __syncthreads();

  // scores[h][i][j] = 0.125 * q_i . k_j
  for (int idx = t; idx < 512; idx += 256) {
    const int h = idx >> 6, rem = idx & 63, i = rem >> 3, j = rem & 7;
    float sc = 0.0f;
    const float* qi = &qkv[i][h * HD];
    const float* kj = &qkv[j][512 + h * HD];
#pragma unroll 8
    for (int d = 0; d < HD; ++d) sc += qi[d] * kj[d];
    scores[h][i][j] = sc * 0.125f;
  }
  __syncthreads();

  if (t < 64) {                       // softmax over j per (h,i)
    const int h = t >> 3, i = t & 7;
    float m = scores[h][i][0];
    for (int j = 1; j < 8; ++j) m = fmaxf(m, scores[h][i][j]);
    float e[8], z = 0.0f;
    for (int j = 0; j < 8; ++j) { e[j] = __expf(scores[h][i][j] - m); z += e[j]; }
    const float inv = 1.0f / z;
    for (int j = 0; j < 8; ++j) scores[h][i][j] = e[j] * inv;
  }
  __syncthreads();

  // asp[s][c] = sum_j attn[h][s][j] * v[j][h*64+d]; stage in registers,
  // then overwrite the (dead) q region of LDS.
  float aspv[16];
#pragma unroll
  for (int ii = 0; ii < 16; ++ii) {
    const int o = t * 16 + ii;
    const int s = o >> 9, c = o & 511, h = c >> 6, d = c & 63;
    float acc = 0.0f;
#pragma unroll
    for (int j = 0; j < 8; ++j)
      acc += scores[h][s][j] * qkv[j][1024 + h * HD + d];
    aspv[ii] = acc;
  }
  __syncthreads();
#pragma unroll
  for (int ii = 0; ii < 16; ++ii) {
    const int o = t * 16 + ii;
    qkv[o >> 9][o & 511] = aspv[ii];
  }
  __syncthreads();

  // asp2[s][c] = sum_{s2} w_cross[s][s2] * asp[s2][c]
  for (int ii = 0; ii < 16; ++ii) {
    const int o = t + 256 * ii;       // coalesced store
    const int s = o >> 9, c = o & 511;
    float acc = 0.0f;
#pragma unroll
    for (int s2 = 0; s2 < 8; ++s2) acc += w_cross[s * 8 + s2] * qkv[s2][c];
    asp2[(size_t)b * (Ss * Cc) + o] = acc;
  }
}

// ---------------------------------------------------------------------------
// K5: out[b,n,c] = LN_c( sum_s A[b][s*4096+n]*asp2[b,s,c] )*gamma+beta + x.
// Block = (b, 64-row tile); one wave32 per row, 16 c's per lane.
// ---------------------------------------------------------------------------
__global__ __launch_bounds__(256) void k5_out(
    const float* __restrict__ x, const float* __restrict__ A,
    const float* __restrict__ asp2, const float* __restrict__ gamma,
    const float* __restrict__ beta, float* __restrict__ out)
{
  const int b = blockIdx.x >> 6;
  const int tile = blockIdx.x & 63;
  const int t = threadIdx.x, lane = t & 31, wave = t >> 5;

  __shared__ float asp[8][512];       // 16 KB
  const float4* aspb4 = (const float4*)(asp2 + (size_t)b * (Ss * Cc));
  for (int i = t; i < (Ss * Cc) / 4; i += 256)
    *(float4*)&asp[i >> 7][(i & 127) * 4] = aspb4[i];
  __syncthreads();

  const float* Ab = A + (size_t)b * (Nn * Ss);
  for (int r = 0; r < 8; ++r) {
    const int n = tile * 64 + wave * 8 + r;
    float a[8];
#pragma unroll
    for (int s = 0; s < 8; ++s) a[s] = Ab[s * 4096 + n];

    float v[16], sum = 0.0f, sq = 0.0f;
#pragma unroll
    for (int j = 0; j < 16; ++j) {
      const int c = j * 32 + lane;
      float acc = 0.0f;
#pragma unroll
      for (int s = 0; s < 8; ++s) acc += a[s] * asp[s][c];
      v[j] = acc; sum += acc; sq += acc * acc;
    }
    for (int off = 16; off > 0; off >>= 1) {
      sum += __shfl_xor(sum, off, 32);
      sq  += __shfl_xor(sq,  off, 32);
    }
    const float mean = sum * (1.0f / Cc);
    const float var  = sq * (1.0f / Cc) - mean * mean;
    const float rstd = rsqrtf(var + 1e-5f);

    const float* xr = x + ((size_t)b * Nn + n) * Cc;
    float* orow = out + ((size_t)b * Nn + n) * Cc;
    __builtin_prefetch(xr + 512, 0, 3);
#pragma unroll
    for (int j = 0; j < 16; ++j) {
      const int c = j * 32 + lane;
      orow[c] = (v[j] - mean) * rstd * gamma[c] + beta[c] + xr[c];
    }
  }
}

// ---------------------------------------------------------------------------
extern "C" void kernel_launch(void* const* d_in, const int* in_sizes, int n_in,
                              void* d_out, int out_size, void* d_ws, size_t ws_size,
                              hipStream_t stream) {
  (void)in_sizes; (void)n_in; (void)out_size; (void)ws_size;
  const float* x       = (const float*)d_in[0];
  const float* w_sum   = (const float*)d_in[1];
  const float* b_sum   = (const float*)d_in[2];
  const float* w_qkv   = (const float*)d_in[3];
  const float* w_cross = (const float*)d_in[4];
  const float* gamma   = (const float*)d_in[5];
  const float* beta    = (const float*)d_in[6];
  float* out = (float*)d_out;

  float* A      = (float*)d_ws;                     // B*N*S   = 524288 f
  float* P      = A + (size_t)Bb * Nn * Ss;         // B*N*S   = 524288 f
  float* tokens = P + (size_t)Bb * Nn * Ss;         // B*S*C   =  65536 f
  float* asp2   = tokens + (size_t)Bb * Ss * Cc;    // B*S*C   =  65536 f

  k1_spaceattn<<<dim3(512), dim3(256), 0, stream>>>(x, w_sum, b_sum, A);
  k2_softmax  <<<dim3(Bb * Ss), dim3(256), 0, stream>>>(A, P);
  k3_tokens   <<<dim3(Bb * 16), dim3(256), 0, stream>>>(x, P, tokens);
  k4_attn     <<<dim3(Bb), dim3(256), 0, stream>>>(tokens, w_qkv, w_cross, asp2);
  k5_out      <<<dim3(Bb * 64), dim3(256), 0, stream>>>(x, A, asp2, gamma, beta, out);
}
